// UniformAggregator_50852412785040
// MI455X (gfx1250) — matrix operand order repaired and checked by reading, not compiled, not run on hardware
//
#include <hip/hip_runtime.h>

typedef __attribute__((ext_vector_type(16))) __bf16 v16bf;
typedef __attribute__((ext_vector_type(8)))  __bf16 v8bf;
typedef __attribute__((ext_vector_type(8)))  float  v8f;

#define FEAT 256
#define K2   512            // 2*FEAT
#define EMB  256
#define NS   10             // neighbors per node
#define TILE_M 64
#define LDS_STRIDE (K2 + 8) // +8 bf16 pad -> conflict-free ds_load_b128

__device__ __forceinline__ unsigned short f2bf(float f) {
    unsigned int u = __float_as_uint(f);
    u += 0x7FFFu + ((u >> 16) & 1u);   // round-to-nearest-even
    return (unsigned short)(u >> 16);
}

// One-shot: W fp32 [256][512] -> bf16 in workspace (stays hot in L2)
__global__ void wprep_kernel(const float* __restrict__ w,
                             unsigned short* __restrict__ wbf, int n) {
    int i = blockIdx.x * blockDim.x + threadIdx.x;
    if (i < n) wbf[i] = f2bf(w[i]);
}

__global__ __launch_bounds__(256)
void sage_wmma_kernel(const int*   __restrict__ self_idx,
                      const int*   __restrict__ neigh_idx,
                      const float* __restrict__ self_table,
                      const float* __restrict__ neigh_table,
                      const unsigned short* __restrict__ wbf,
                      float* __restrict__ out, int B)
{
    __shared__ unsigned short sComb[TILE_M * LDS_STRIDE]; // [64][520] bf16
    __shared__ int sNIdx[TILE_M * NS];
    __shared__ int sSIdx[TILE_M];

    const int tid    = threadIdx.x;
    const int wgBase = blockIdx.x * TILE_M;

    // ---- stage indices (clamp OOB rows to 0; their stores are guarded later)
    if (tid < TILE_M) {
        int r = wgBase + tid;
        sSIdx[tid] = (r < B) ? self_idx[r] : 0;
    }
    for (int i = tid; i < TILE_M * NS; i += 256) {
        int r = wgBase + i / NS;
        sNIdx[i] = (r < B) ? neigh_idx[r * NS + (i % NS)] : 0;
    }
    __syncthreads();

    // ---- gather + mean: 64 threads per row (float4 each), 4 rows per pass
    {
        const int t    = tid & 63;   // feature group: floats [4t, 4t+4)
        const int rsub = tid >> 6;   // 0..3
        const float inv = 1.0f / (float)NS;
        for (int pass = 0; pass < TILE_M / 4; ++pass) {
            const int lr = pass * 4 + rsub;
            const int f  = t * 4;
            // self features -> combined[:, 0:256)
            {
                const float4 v = *(const float4*)(self_table + (long)sSIdx[lr] * FEAT + f);
                unsigned short* dst = &sComb[lr * LDS_STRIDE + f];
                *(unsigned int*)(dst)     = (unsigned)f2bf(v.x) | ((unsigned)f2bf(v.y) << 16);
                *(unsigned int*)(dst + 2) = (unsigned)f2bf(v.z) | ((unsigned)f2bf(v.w) << 16);
            }
            // neighbor mean -> combined[:, 256:512)
            {
                float ax = 0.f, ay = 0.f, az = 0.f, aw = 0.f;
                #pragma unroll
                for (int s = 0; s < NS; ++s) {
                    const float4 v = *(const float4*)(neigh_table +
                                       (long)sNIdx[lr * NS + s] * FEAT + f);
                    ax += v.x; ay += v.y; az += v.z; aw += v.w;
                }
                unsigned short* dst = &sComb[lr * LDS_STRIDE + FEAT + f];
                *(unsigned int*)(dst)     = (unsigned)f2bf(ax * inv) | ((unsigned)f2bf(ay * inv) << 16);
                *(unsigned int*)(dst + 2) = (unsigned)f2bf(az * inv) | ((unsigned)f2bf(aw * inv) << 16);
            }
        }
    }
    __syncthreads();

    // ---- WMMA GEMM: out[64,256] = combined[64,512] @ W^T  (bf16 in, f32 acc)
    const int lane  = tid & 31;
    const int wave  = tid >> 5;     // 0..7
    const int waveM = wave >> 2;    // 0..1  -> 32 rows each
    const int waveN = wave & 3;     // 0..3  -> 64 cols each
    const int lrow  = lane & 15;
    const int khalf = lane >> 4;    // 0/1 selects K half per ISA fragment layout

    v8f acc[2][4];
    #pragma unroll
    for (int mf = 0; mf < 2; ++mf)
        #pragma unroll
        for (int nf = 0; nf < 4; ++nf)
            acc[mf][nf] = (v8f)(0.0f);

    for (int k0 = 0; k0 < K2; k0 += 32) {
        // A fragments from LDS: lane holds M=lrow, K = k0 + 8*khalf + [0..8) and +16
        v16bf a[2];
        #pragma unroll
        for (int mf = 0; mf < 2; ++mf) {
            const int row = waveM * 32 + mf * 16 + lrow;
            const unsigned short* p = &sComb[row * LDS_STRIDE + k0 + khalf * 8];
            union { v16bf v; v8bf h[2]; } u;
            u.h[0] = *(const v8bf*)(p);
            u.h[1] = *(const v8bf*)(p + 16);
            a[mf] = u.v;
        }
        #pragma unroll
        for (int nf = 0; nf < 4; ++nf) {
            // B fragment from bf16 W (L2-resident): lane holds N=lrow,
            // K = k0 + 8*khalf + [0..8) and +16  (B[k][n] = W[n][k])
            const int n = waveN * 64 + nf * 16 + lrow;
            const unsigned short* p = wbf + (long)n * K2 + k0 + khalf * 8;
            union { v16bf v; v8bf h[2]; } u;
            u.h[0] = *(const v8bf*)(p);
            u.h[1] = *(const v8bf*)(p + 16);
            const v16bf b = u.v;
            #pragma unroll
            for (int mf = 0; mf < 2; ++mf) {
                acc[mf][nf] = __builtin_amdgcn_wmma_f32_16x16x32_bf16(
                    false, a[mf], false, b, (short)0, acc[mf][nf], false, false);
            }
        }
    }

    // ---- epilogue: ReLU + store (C layout: VGPR r -> M=r (+8 for lanes>=16), N=lane&15)
    #pragma unroll
    for (int mf = 0; mf < 2; ++mf) {
        #pragma unroll
        for (int nf = 0; nf < 4; ++nf) {
            const int col = waveN * 64 + nf * 16 + lrow;
            #pragma unroll
            for (int r = 0; r < 8; ++r) {
                const int row = wgBase + waveM * 32 + mf * 16 + r + khalf * 8;
                if (row < B) {
                    float v = acc[mf][nf][r];
                    out[(long)row * EMB + col] = v > 0.f ? v : 0.f;
                }
            }
        }
    }
}

extern "C" void kernel_launch(void* const* d_in, const int* in_sizes, int n_in,
                              void* d_out, int out_size, void* d_ws, size_t ws_size,
                              hipStream_t stream) {
    const int*   self_idx    = (const int*)  d_in[0];
    const int*   neigh_idx   = (const int*)  d_in[1];
    const float* self_table  = (const float*)d_in[2];
    const float* neigh_table = (const float*)d_in[3];
    const float* weight      = (const float*)d_in[4];
    float* out = (float*)d_out;
    const int B = in_sizes[0];
    const int wElems = in_sizes[4];             // 256*512
    unsigned short* wbf = (unsigned short*)d_ws; // 256 KB bf16 W

    wprep_kernel<<<(wElems + 255) / 256, 256, 0, stream>>>(weight, wbf, wElems);

    const int grid = (B + TILE_M - 1) / TILE_M;
    sage_wmma_kernel<<<grid, 256, 0, stream>>>(self_idx, neigh_idx, self_table,
                                               neigh_table, wbf, out, B);
}
